// DiagGateRNN_46136538694215
// MI455X (gfx1250) — matrix operand order repaired and checked by reading, not compiled, not run on hardware
//
#include <hip/hip_runtime.h>

// ---------------------------------------------------------------------------
// DiagGateRNN for MI455X (gfx1250): WMMA bf16 persistent-scan implementation
//   B=64, T=512, D=128, H=512
// Round 2: stage the hidden state into LDS each step with
// GLOBAL_LOAD_ASYNC_TO_LDS_B128 (ASYNCcnt path) instead of per-wave global
// fragment loads.
// ---------------------------------------------------------------------------

typedef __attribute__((ext_vector_type(16))) __bf16 v16bf;
typedef __attribute__((ext_vector_type(8)))  float  v8f;

union FragAB { v16bf v; unsigned int u[8]; };

static constexpr int  Bb = 64, Tt = 512, Dd = 128, Hh = 512;
static constexpr int  Mrows = Bb * Tt;                 // 32768
// d_out float offsets: ys | hs | ss | leaks | rdiags
static constexpr long OFF_YS = 0;
static constexpr long OFF_HS = 32768;
static constexpr long OFF_SS = OFF_HS + (long)Mrows * Hh;   // 16,809,984
static constexpr long OFF_LK = OFF_SS + (long)Mrows * Hh;   // 33,587,200
static constexpr long OFF_RD = OFF_LK + (long)Mrows * Hh;   // 50,364,416
// workspace byte offsets
static constexpr size_t WS_HBF0 = 0;          // 64 KB  bf16 h (buf 0)
static constexpr size_t WS_HBF1 = 65536;      // 64 KB  bf16 h (buf 1)
static constexpr size_t WS_HF0  = 131072;     // 128 KB f32 h (buf 0)
static constexpr size_t WS_HF1  = 262144;     // 128 KB f32 h (buf 1)
static constexpr size_t WS_BAR  = 393216;     // 2 KB   per-step barrier counters
static constexpr size_t WS_TOTAL = WS_BAR + 2048;

__device__ inline unsigned short f32_to_bf16(float f) {
    union { float f; unsigned int u; } c; c.f = f;
    unsigned int r = c.u + 0x7FFFu + ((c.u >> 16) & 1u);  // round-nearest-even
    return (unsigned short)(r >> 16);
}

// byte address of a __shared__ object inside the LDS aperture (for async ops)
__device__ inline unsigned lds_addr(const void* p) {
    return (unsigned)(unsigned long long)
        (const __attribute__((address_space(3))) void*)p;
}

// ---------------------------------------------------------------------------
// Kernel A: xp = x*Wx^T + bx   -> stored into hs region
//           sp = x*Ws^T + bs   -> stored into ss region
// One block = 16 M-rows x 128 N-cols; 8 waves, one 16x16 WMMA tile each.
// ---------------------------------------------------------------------------
__global__ __launch_bounds__(256) void proj_kernel(
    const float* __restrict__ x,
    const float* __restrict__ Wx, const float* __restrict__ Wxb,
    const float* __restrict__ Ws, const float* __restrict__ Wsb,
    float* __restrict__ out)
{
    __shared__ unsigned short sA[16 * 128];     // 4 KB  x slab (bf16)
    __shared__ unsigned short sB[128 * 128];    // 32 KB weight slab (bf16)

    const int tid = threadIdx.x;
    const int bx  = blockIdx.x;                 // M tile (16 rows)
    const int ng  = blockIdx.y;                 // N group (128 of 1024 fused cols)

    for (int i = tid; i < 16 * 128; i += 256) {
        int m = i >> 7, k = i & 127;
        sA[i] = f32_to_bf16(x[(bx * 16 + m) * Dd + k]);
    }
    for (int i = tid; i < 128 * 128; i += 256) {
        int nl = i >> 7, k = i & 127;
        int n = ng * 128 + nl;
        float w = (n < Hh) ? Wx[n * Dd + k] : Ws[(n - Hh) * Dd + k];
        sB[i] = f32_to_bf16(w);
    }
    // warm next M slab through GL2 (global_prefetch_b8)
    __builtin_prefetch(x + ((size_t)bx * 16 + 16) * Dd, 0, 1);
    __syncthreads();

    const int wv = tid >> 5, lane = tid & 31;
    const int hi = lane >> 4, lo = lane & 15;
    const int nl0 = wv * 16;

    v8f acc = {};
    #pragma unroll
    for (int kc = 0; kc < 4; ++kc) {
        const int kb = kc * 32;
        FragAB a, b;
        #pragma unroll
        for (int j = 0; j < 8; ++j) {
            int ka = kb + ((j < 4) ? (2 * j) : (16 + 2 * (j - 4))) + 8 * hi;
            a.u[j] = *(const unsigned int*)&sA[lo * 128 + ka];
            int kv = kb + ((j < 4) ? (2 * j) : (16 + 2 * (j - 4))) + 8 * hi;
            b.u[j] = *(const unsigned int*)&sB[(nl0 + lo) * 128 + kv];
        }
        acc = __builtin_amdgcn_wmma_f32_16x16x32_bf16(
            false, a.v, false, b.v, (short)0, acc, false, false);
    }

    const int  n    = ng * 128 + nl0 + lo;
    const float bias = (n < Hh) ? Wxb[n] : Wsb[n - Hh];
    const long base  = (n < Hh) ? (OFF_HS + n) : (OFF_SS + (n - Hh));
    #pragma unroll
    for (int r = 0; r < 8; ++r) {
        long row = (long)bx * 16 + r + 8 * hi;
        out[base + row * Hh] = acc[r] + bias;
    }
}

// ---------------------------------------------------------------------------
// Kernel B: persistent scan. 16 blocks; block owns H columns [blk*32, blk*32+32).
// Wh/Us slices live in LDS (bf16, 64 KB); the bf16 hidden state (64 KB) is
// asynchronously staged global->LDS each step via GLOBAL_LOAD_ASYNC_TO_LDS_B128
// so both WMMA operands are fed from LDS. h double-buffered in global
// (L2-resident). Per-step counter barrier across the 16 blocks.
// ---------------------------------------------------------------------------
__global__ __launch_bounds__(256) void scan_kernel(
    const float* __restrict__ Wh, const float* __restrict__ Us,
    float* __restrict__ out, unsigned char* __restrict__ ws)
{
    __shared__ unsigned short sWh[32 * 512];    // 32 KB
    __shared__ unsigned short sUs[32 * 512];    // 32 KB
    __shared__ unsigned short sH [64 * 512];    // 64 KB bf16 h, staged per step

    const int tid = threadIdx.x;
    const int blk = blockIdx.x;
    const int nb  = blk * 32;

    unsigned short* hbf0 = (unsigned short*)(ws + WS_HBF0);
    unsigned short* hbf1 = (unsigned short*)(ws + WS_HBF1);
    float*          hf0  = (float*)(ws + WS_HF0);
    float*          hf1  = (float*)(ws + WS_HF1);
    unsigned int*   bar  = (unsigned int*)(ws + WS_BAR);

    for (int i = tid; i < 32 * 512; i += 256) {
        int nl = i >> 9, k = i & 511;
        sWh[i] = f32_to_bf16(Wh[(nb + nl) * Hh + k]);
        sUs[i] = f32_to_bf16(Us[(nb + nl) * Hh + k]);
    }

    const int wv = tid >> 5, lane = tid & 31;
    const int hi = lane >> 4, lo = lane & 15;
    const int mt = wv >> 1, nt = wv & 1;        // 4 M-tiles x 2 N-tiles
    const int nLoc = nt * 16 + lo;              // 0..31
    const int nG   = nb + nLoc;                 // global H column
    const float wh_d = Wh[nG * Hh + nG];        // diagonal terms (f32, L2 hot)
    const float us_d = Us[nG * Hh + nG];
    const unsigned sH_base = lds_addr(&sH[0]);
    __syncthreads();

    for (int t = 0; t < Tt; ++t) {
        const int cur = t & 1;
        const unsigned short* hc = cur ? hbf1 : hbf0;
        unsigned short*       hn_bf = cur ? hbf0 : hbf1;
        const float*          hfc = cur ? hf1 : hf0;
        float*                hfn = cur ? hf0 : hf1;

        // ---- async stage: 64 KB bf16 h  global -> LDS  (4096 x B128) ----
        for (int i = tid; i < 4096; i += 256) {
            unsigned long long ga =
                (unsigned long long)(const void*)(hc + (size_t)i * 8);
            unsigned loff = sH_base + (unsigned)i * 16u;
            asm volatile("global_load_async_to_lds_b128 %0, %1, off"
                         :: "v"(loff), "v"(ga) : "memory");
        }
        asm volatile("s_wait_asynccnt 0x0" ::: "memory");
        __syncthreads();

        v8f pre = {}, gpre = {};
        #pragma unroll 4
        for (int kc = 0; kc < 16; ++kc) {
            const int kb = kc * 32;
            FragAB a, bw, bu;
            #pragma unroll
            for (int j = 0; j < 8; ++j) {
                int ka = kb + ((j < 4) ? (2 * j) : (16 + 2 * (j - 4))) + 8 * hi;
                a.u[j] = *(const unsigned int*)&sH[(mt * 16 + lo) * Hh + ka];
                int kv = kb + ((j < 4) ? (2 * j) : (16 + 2 * (j - 4))) + 8 * hi;
                bw.u[j] = *(const unsigned int*)&sWh[nLoc * Hh + kv];
                bu.u[j] = *(const unsigned int*)&sUs[nLoc * Hh + kv];
            }
            pre  = __builtin_amdgcn_wmma_f32_16x16x32_bf16(
                       false, a.v, false, bw.v, (short)0, pre,  false, false);
            gpre = __builtin_amdgcn_wmma_f32_16x16x32_bf16(
                       false, a.v, false, bu.v, (short)0, gpre, false, false);
        }

        #pragma unroll
        for (int r = 0; r < 8; ++r) {
            const int  b   = mt * 16 + r + 8 * hi;
            const long row = (long)b * Tt + t;
            const long e   = row * Hh + nG;
            const float xp    = out[OFF_HS + e];      // pre-staged projection
            const float sp    = out[OFF_SS + e];
            const float h_old = hfc[b * Hh + nG];
            const float s  = 1.0f / (1.0f + __expf(-(sp + gpre[r])));
            const float z  = __expf(2.0f * (xp + pre[r]));
            const float ht = (z - 1.0f) / (z + 1.0f);  // tanh
            const float hv = (1.0f - s) * h_old + s * ht;
            const float rd = (ht - h_old) * (s * (1.0f - s) * us_d)
                           + s * (1.0f - ht * ht) * wh_d;
            out[OFF_HS + e] = hv;                      // overwrites xp (done with it)
            out[OFF_SS + e] = s;                       // overwrites sp
            out[OFF_LK + e] = 1.0f - s;
            out[OFF_RD + e] = rd;
            hfn[b * Hh + nG]   = hv;
            hn_bf[b * Hh + nG] = f32_to_bf16(hv);
        }

        // --- grid barrier over the 16 persistent blocks -------------------
        __threadfence();
        __syncthreads();
        if (tid == 0) {
            atomicAdd(&bar[t], 1u);
            while (atomicAdd(&bar[t], 0u) < (unsigned)gridDim.x)
                __builtin_amdgcn_s_sleep(1);
        }
        __syncthreads();
        __threadfence();
    }
}

// ---------------------------------------------------------------------------
// Kernel C: ys[b,t] = hs[b,t,:] . out_w + out_b   (one wave per row)
// ---------------------------------------------------------------------------
__global__ __launch_bounds__(256) void yout_kernel(
    const float* __restrict__ ow, const float* __restrict__ ob,
    float* __restrict__ out)
{
    const int wv = threadIdx.x >> 5, lane = threadIdx.x & 31;
    const long row = (long)blockIdx.x * 8 + wv;
    const float* hs = out + OFF_HS + row * Hh;
    float sum = 0.0f;
    #pragma unroll
    for (int i = lane; i < Hh; i += 32) sum += hs[i] * ow[i];
    #pragma unroll
    for (int o = 16; o > 0; o >>= 1) sum += __shfl_xor(sum, o, 32);
    if (lane == 0) out[OFF_YS + row] = sum + ob[0];
}

// ---------------------------------------------------------------------------
extern "C" void kernel_launch(void* const* d_in, const int* in_sizes, int n_in,
                              void* d_out, int out_size, void* d_ws, size_t ws_size,
                              hipStream_t stream)
{
    const float* x   = (const float*)d_in[0];
    const float* Wx  = (const float*)d_in[1];
    const float* Wxb = (const float*)d_in[2];
    const float* Wh  = (const float*)d_in[3];
    const float* Ws  = (const float*)d_in[4];
    const float* Wsb = (const float*)d_in[5];
    const float* Us  = (const float*)d_in[6];
    const float* ow  = (const float*)d_in[7];
    const float* ob  = (const float*)d_in[8];
    float* out = (float*)d_out;

    // zero h0 (both buffers) + barrier counters; capture-safe, deterministic
    hipMemsetAsync(d_ws, 0, WS_TOTAL, stream);

    proj_kernel<<<dim3(Mrows / 16, 8), 256, 0, stream>>>(x, Wx, Wxb, Ws, Wsb, out);
    scan_kernel<<<dim3(Hh / 32), 256, 0, stream>>>(Wh, Us, out, (unsigned char*)d_ws);
    yout_kernel<<<dim3(Mrows / 8), 256, 0, stream>>>(ow, ob, out);
}
